// TAGCN_49289044689243
// MI455X (gfx1250) — compile-verified, matrix operand order checked by reading, not run on hardware
//
#include <hip/hip_runtime.h>

typedef __attribute__((ext_vector_type(16))) __bf16 v16bf;
typedef __attribute__((ext_vector_type(8)))  __bf16 v8bf;
typedef __attribute__((ext_vector_type(8)))  float  v8f;

// ---------------------------------------------------------------- utilities

__global__ void fill_zero4_kernel(float4* __restrict__ p, int n4) {
    int t = blockIdx.x * blockDim.x + threadIdx.x;
    if (t < n4) p[t] = make_float4(0.f, 0.f, 0.f, 0.f);
}

__global__ void deg_kernel(const long long* __restrict__ ei, float* __restrict__ deg, int E) {
    int e = blockIdx.x * blockDim.x + threadIdx.x;
    if (e < E) atomicAdd(deg + (int)ei[(long long)E + e], 1.0f);
}

__global__ void dinv_kernel(float* __restrict__ d, int N) {
    int n = blockIdx.x * blockDim.x + threadIdx.x;
    if (n < N) {
        float v = d[n];
        d[n] = (v > 0.f) ? rsqrtf(fmaxf(v, 1.f)) : 0.f;
    }
}

__global__ void norm_kernel(const long long* __restrict__ ei, const float* __restrict__ dinv,
                            float* __restrict__ nrm, int E) {
    int e = blockIdx.x * blockDim.x + threadIdx.x;
    if (e < E) nrm[e] = dinv[(int)ei[e]] * dinv[(int)ei[(long long)E + e]];
}

// W [K+1][Cin][Cout] f32  ->  Wt [K+1][Cout][Cin] bf16 (contiguous K runs per column)
__global__ void wtrans_kernel(const float* __restrict__ W, __bf16* __restrict__ Wt,
                              int Cin, int Cout, int total) {
    int t = blockIdx.x * blockDim.x + threadIdx.x;
    if (t >= total) return;
    int per = Cin * Cout;
    int k = t / per, r = t % per;
    int c = r / Cout, n = r % Cout;
    Wt[k * per + n * Cin + c] = (__bf16)W[t];
}

__global__ void bias_init_kernel(float* __restrict__ acc, const float* __restrict__ b,
                                 int total, int C) {
    int t = blockIdx.x * blockDim.x + threadIdx.x;
    if (t < total) acc[t] = b[t & (C - 1)];   // C is a power of two (256, 64)
}

__global__ void relu_kernel(float* __restrict__ a, int n) {
    int t = blockIdx.x * blockDim.x + threadIdx.x;
    if (t < n) a[t] = fmaxf(a[t], 0.f);
}

// ------------------------------------------------- edge-parallel propagation
// one wave (32 lanes) per edge: coalesced float4 gather of h[row], f32 atomic
// scatter into h[col]; norm applied per edge.
template <int C>
__global__ void spmm_kernel(const float* __restrict__ hin, float* __restrict__ hout,
                            const long long* __restrict__ ei, const float* __restrict__ nrm,
                            int E) {
    int w = (blockIdx.x * blockDim.x + threadIdx.x) >> 5;   // wave-uniform
    if (w >= E) return;
    int lane = threadIdx.x & 31;
    int r = (int)ei[w];
    int c = (int)ei[(long long)E + w];
    float nw = nrm[w];
    const float4* src = (const float4*)(hin + (long long)r * C);
    float*       dst  = hout + (long long)c * C;
#pragma unroll
    for (int i = 0; i < C / 128; ++i) {
        float4 v = src[i * 32 + lane];
        int b = (i * 32 + lane) * 4;
        atomicAdd(dst + b + 0, nw * v.x);
        atomicAdd(dst + b + 1, nw * v.y);
        atomicAdd(dst + b + 2, nw * v.z);
        atomicAdd(dst + b + 3, nw * v.w);
    }
}

// --------------------------------------------------- bf16 WMMA GEMM (accum)
// acc[N x COUT] += H[N x CIN] (f32, cvt->bf16) @ Wt^T  (Wt is [COUT x CIN] bf16)
// One wave computes a 16x16 tile, K stepped by 32 via v_wmma_f32_16x16x32_bf16.
template <int CIN, int COUT>
__global__ void __launch_bounds__(256)
gemm_acc_kernel(const float* __restrict__ H, const __bf16* __restrict__ Wt,
                float* __restrict__ acc, int nTiles, int tilesN) {
    int w = (blockIdx.x * blockDim.x + threadIdx.x) >> 5;   // wave-uniform
    if (w >= nTiles) return;                                // full-EXEC waves proceed
    int lane = threadIdx.x & 31;
    int tN = w % tilesN;
    int tM = w / tilesN;

    int nl = lane & 15;
    int hi = lane >> 4;         // half-wave select
    int mb = hi * 8;            // C/D fragment: rows mb..mb+7 at column nl
    float* accp = acc + (long long)(tM * 16 + mb) * COUT + tN * 16 + nl;
    v8f c;
#pragma unroll
    for (int v = 0; v < 8; ++v) c[v] = accp[(long long)v * COUT];

    int g0 = hi * 8;            // A/B fragment K sub-offset (0 or 8)
    const float*  hrow = H  + (long long)(tM * 16 + nl) * CIN;
    const __bf16* wrow = Wt + (long long)(tN * 16 + nl) * CIN;

#pragma unroll
    for (int k0 = 0; k0 < CIN; k0 += 32) {
        const float4* pa0 = (const float4*)(hrow + k0 + g0);
        const float4* pa1 = (const float4*)(hrow + k0 + g0 + 16);
        float4 f0 = pa0[0], f1 = pa0[1];      // K = g0 .. g0+7
        float4 f2 = pa1[0], f3 = pa1[1];      // K = g0+16 .. g0+23
        v16bf a;
        a[0]  = (__bf16)f0.x; a[1]  = (__bf16)f0.y; a[2]  = (__bf16)f0.z; a[3]  = (__bf16)f0.w;
        a[4]  = (__bf16)f1.x; a[5]  = (__bf16)f1.y; a[6]  = (__bf16)f1.z; a[7]  = (__bf16)f1.w;
        a[8]  = (__bf16)f2.x; a[9]  = (__bf16)f2.y; a[10] = (__bf16)f2.z; a[11] = (__bf16)f2.w;
        a[12] = (__bf16)f3.x; a[13] = (__bf16)f3.y; a[14] = (__bf16)f3.z; a[15] = (__bf16)f3.w;

        v8bf b0 = *(const v8bf*)(wrow + k0 + g0);        // 16B contiguous
        v8bf b1 = *(const v8bf*)(wrow + k0 + g0 + 16);
        v16bf b;
#pragma unroll
        for (int i = 0; i < 8; ++i) { b[i] = b0[i]; b[8 + i] = b1[i]; }

        c = __builtin_amdgcn_wmma_f32_16x16x32_bf16(
                /*neg_a=*/false, a, /*neg_b=*/false, b,
                /*c_mod=*/(short)0, c, /*reuse_a=*/false, /*reuse_b=*/false);
    }
#pragma unroll
    for (int v = 0; v < 8; ++v) accp[(long long)v * COUT] = c[v];
}

// --------------------------------------------------------------- log_softmax
// one wave per node, 64 channels = 2 floats/lane, butterfly reductions (wave32)
__global__ void lsm_kernel(const float* __restrict__ acc, float* __restrict__ out, int N) {
    int w = (blockIdx.x * blockDim.x + threadIdx.x) >> 5;
    if (w >= N) return;
    int lane = threadIdx.x & 31;
    float2 v = ((const float2*)(acc + (long long)w * 64))[lane];
    float m = fmaxf(v.x, v.y);
#pragma unroll
    for (int off = 16; off; off >>= 1) m = fmaxf(m, __shfl_xor(m, off, 32));
    float s = expf(v.x - m) + expf(v.y - m);
#pragma unroll
    for (int off = 16; off; off >>= 1) s += __shfl_xor(s, off, 32);
    float l = m + logf(s);
    ((float2*)(out + (long long)w * 64))[lane] = make_float2(v.x - l, v.y - l);
}

// -------------------------------------------------------------------- launch

static inline int cdiv(long long a, long long b) { return (int)((a + b - 1) / b); }

extern "C" void kernel_launch(void* const* d_in, const int* in_sizes, int n_in,
                              void* d_out, int out_size, void* d_ws, size_t ws_size,
                              hipStream_t stream) {
    const float*     x  = (const float*)d_in[0];
    const long long* ei = (const long long*)d_in[1];   // int64 edge_index [2, E]
    const float*     W1 = (const float*)d_in[2];
    const float*     b1 = (const float*)d_in[3];
    const float*     W2 = (const float*)d_in[4];
    const float*     b2 = (const float*)d_in[5];
    float*           out = (float*)d_out;

    const int C0 = 128, C1 = 256, C2 = 64, KH = 4;     // KH = K+1 hop weights
    int N = in_sizes[0] / C0;
    int E = in_sizes[1] / 2;

    // ---- carve workspace (256B aligned)
    size_t off = 0;
    auto carve = [&](size_t bytes) {
        void* p = (char*)d_ws + off;
        off += (bytes + 255) & ~(size_t)255;
        return p;
    };
    float*  dinv = (float*)carve((size_t)N * 4);
    float*  nrm  = (float*)carve((size_t)E * 4);
    __bf16* w1t  = (__bf16*)carve((size_t)KH * C1 * C0 * 2);
    __bf16* w2t  = (__bf16*)carve((size_t)KH * C2 * C1 * 2);
    float*  reg1 = (float*)carve((size_t)N * C1 * 4);  // layer1: bufA|bufB, layer2: bufC
    float*  reg2 = (float*)carve((size_t)N * C1 * 4);  // layer2: bufD
    float*  acc1 = (float*)carve((size_t)N * C1 * 4);  // layer1 accum, relu'd in place
    float*  acc2 = (float*)carve((size_t)N * C2 * 4);  // layer2 accum
    if (off > ws_size) return;

    float* bufA = reg1;
    float* bufB = reg1 + (size_t)N * C0;
    float* bufC = reg1;
    float* bufD = reg2;

    const int TB = 256;

    // ---- gcn_norm
    fill_zero4_kernel<<<cdiv(N / 4, TB), TB, 0, stream>>>((float4*)dinv, N / 4);
    deg_kernel<<<cdiv(E, TB), TB, 0, stream>>>(ei, dinv, E);
    dinv_kernel<<<cdiv(N, TB), TB, 0, stream>>>(dinv, N);
    norm_kernel<<<cdiv(E, TB), TB, 0, stream>>>(ei, dinv, nrm, E);

    // ---- weights -> bf16, transposed [Cout x Cin]
    wtrans_kernel<<<cdiv(KH * C0 * C1, TB), TB, 0, stream>>>(W1, w1t, C0, C1, KH * C0 * C1);
    wtrans_kernel<<<cdiv(KH * C1 * C2, TB), TB, 0, stream>>>(W2, w2t, C1, C2, KH * C1 * C2);

    // ---- layer 1: acc1 = b1 + sum_k (A^k x) W1[k]
    int tiles1 = (N / 16) * (C1 / 16);
    int g1 = cdiv((long long)tiles1 * 32, TB);
    int gE = cdiv((long long)E * 32, TB);

    bias_init_kernel<<<cdiv(N * C1, TB), TB, 0, stream>>>(acc1, b1, N * C1, C1);
    gemm_acc_kernel<128, 256><<<g1, TB, 0, stream>>>(x, w1t, acc1, tiles1, C1 / 16);

    fill_zero4_kernel<<<cdiv((long long)N * C0 / 4, TB), TB, 0, stream>>>((float4*)bufA, N * C0 / 4);
    spmm_kernel<128><<<gE, TB, 0, stream>>>(x, bufA, ei, nrm, E);
    gemm_acc_kernel<128, 256><<<g1, TB, 0, stream>>>(bufA, w1t + 1 * C1 * C0, acc1, tiles1, C1 / 16);

    fill_zero4_kernel<<<cdiv((long long)N * C0 / 4, TB), TB, 0, stream>>>((float4*)bufB, N * C0 / 4);
    spmm_kernel<128><<<gE, TB, 0, stream>>>(bufA, bufB, ei, nrm, E);
    gemm_acc_kernel<128, 256><<<g1, TB, 0, stream>>>(bufB, w1t + 2 * C1 * C0, acc1, tiles1, C1 / 16);

    fill_zero4_kernel<<<cdiv((long long)N * C0 / 4, TB), TB, 0, stream>>>((float4*)bufA, N * C0 / 4);
    spmm_kernel<128><<<gE, TB, 0, stream>>>(bufB, bufA, ei, nrm, E);
    gemm_acc_kernel<128, 256><<<g1, TB, 0, stream>>>(bufA, w1t + 3 * C1 * C0, acc1, tiles1, C1 / 16);

    relu_kernel<<<cdiv(N * C1, TB), TB, 0, stream>>>(acc1, N * C1);   // h1 = relu(acc1)

    // ---- layer 2: acc2 = b2 + sum_k (A^k h1) W2[k]
    int tiles2 = (N / 16) * (C2 / 16);
    int g2 = cdiv((long long)tiles2 * 32, TB);

    bias_init_kernel<<<cdiv(N * C2, TB), TB, 0, stream>>>(acc2, b2, N * C2, C2);
    gemm_acc_kernel<256, 64><<<g2, TB, 0, stream>>>(acc1, w2t, acc2, tiles2, C2 / 16);

    fill_zero4_kernel<<<cdiv((long long)N * C1 / 4, TB), TB, 0, stream>>>((float4*)bufC, N * C1 / 4);
    spmm_kernel<256><<<gE, TB, 0, stream>>>(acc1, bufC, ei, nrm, E);
    gemm_acc_kernel<256, 64><<<g2, TB, 0, stream>>>(bufC, w2t + 1 * C2 * C1, acc2, tiles2, C2 / 16);

    fill_zero4_kernel<<<cdiv((long long)N * C1 / 4, TB), TB, 0, stream>>>((float4*)bufD, N * C1 / 4);
    spmm_kernel<256><<<gE, TB, 0, stream>>>(bufC, bufD, ei, nrm, E);
    gemm_acc_kernel<256, 64><<<g2, TB, 0, stream>>>(bufD, w2t + 2 * C2 * C1, acc2, tiles2, C2 / 16);

    fill_zero4_kernel<<<cdiv((long long)N * C1 / 4, TB), TB, 0, stream>>>((float4*)bufC, N * C1 / 4);
    spmm_kernel<256><<<gE, TB, 0, stream>>>(bufD, bufC, ei, nrm, E);
    gemm_acc_kernel<256, 64><<<g2, TB, 0, stream>>>(bufC, w2t + 3 * C2 * C1, acc2, tiles2, C2 / 16);

    // ---- log_softmax over 64 channels
    lsm_kernel<<<cdiv((long long)N * 32, TB), TB, 0, stream>>>(acc2, out, N);
}
